// DiscreteLSTMActor_61821759259538
// MI455X (gfx1250) — matrix operand order repaired and checked by reading, not compiled, not run on hardware
//
#include <hip/hip_runtime.h>
#include <hip/hip_bf16.h>

// ---------------------------------------------------------------------------
// DiscreteLSTMActor forward for MI455X (gfx1250, wave32, WMMA).
//
// All GEMMs use v_wmma_f32_16x16x32_f16 (f16 in, f32 accumulate).
// K-dim H=528 is padded to HP=544 (17 * 32). Weights are repacked to f16
// row-major [rows][HP] once per call so both A and B WMMA operands can be
// staged into LDS with contiguous 16B transfers via the CDNA5 async
// global->LDS path (GLOBAL_LOAD_ASYNC_TO_LDS_B128 + ASYNCcnt).
// The sequential LSTM step kernel double-buffers its LDS tiles: compute
// from buf[s&1] while async-staging buf[s&1^1] -> one barrier per K-step.
// ---------------------------------------------------------------------------

typedef _Float16 h16;
typedef __attribute__((ext_vector_type(16))) _Float16 v16h;
typedef __attribute__((ext_vector_type(8)))  _Float16 v8h;
typedef __attribute__((ext_vector_type(8)))  float    v8f;
typedef __attribute__((ext_vector_type(4)))  float    v4f;

#define T_DIM  256
#define B_DIM  256
#define OBS_D  512
#define FEAT_D 512
#define NA     15
#define H_D    528
#define HP     544          // padded hidden (17*32)
#define G_D    2112         // 4*H
#define LDA    40           // LDS row stride in halves (80B -> conflict-free)

#define TBROWS (T_DIM * B_DIM)

// d_out layout (floats): policy_logits | baseline | action | hT | cT
#define OFF_BASE 983040u           // 65536*15
#define OFF_ACT  1048576u
#define OFF_HT   1114112u
#define OFF_CT   1384448u          // OFF_HT + 2*256*528

// --------------------------- WMMA helpers ----------------------------------

static __device__ __forceinline__ v8f wmma_f16(v16h a, v16h b, v8f c) {
  // (neg_a, A, neg_b, B, c_mod, C, reuse_a, reuse_b)
  return __builtin_amdgcn_wmma_f32_16x16x32_f16(false, a, false, b, (short)0, c,
                                                false, false);
}

// A fragment (16x32 f16): lane half h -> frag[0..7]=K(h*8+0..7),
// frag[8..15]=K(16+h*8+0..7). LDS tile: [rows][LDA] halves.
static __device__ __forceinline__ v16h frag_a(const h16* s, int rowbase, int lane) {
  const int r = rowbase + (lane & 15);
  const int hs = lane >> 4;
  v8h lo = *(const v8h*)(s + r * LDA + hs * 8);
  v8h hi = *(const v8h*)(s + r * LDA + 16 + hs * 8);
  return __builtin_shufflevector(lo, hi, 0,1,2,3,4,5,6,7,8,9,10,11,12,13,14,15);
}

// B fragment (32x16 f16): lane half h -> frag[0..7]=K(h*8+0..7), frag[8..15]=K(16+h*8+0..7) for col lane&15.
// LDS tile stored K-contiguous per column: [cols][LDA] halves.
static __device__ __forceinline__ v16h frag_b(const h16* s, int colbase, int lane) {
  const int c = colbase + (lane & 15);
  const int hs = lane >> 4;
  v8h lo = *(const v8h*)(s + c * LDA + hs * 8);
  v8h hi = *(const v8h*)(s + c * LDA + 16 + hs * 8);
  return __builtin_shufflevector(lo, hi, 0,1,2,3,4,5,6,7,8,9,10,11,12,13,14,15);
}

// ------------------- CDNA5 async global->LDS staging ------------------------
// Generic pointers to LDS carry the LDS byte offset in their low 32 bits
// (ISA: LDS aperture address calc truncates to addr[31:0]).

static __device__ __forceinline__ void async_cp16(void* lds_dst, const void* gsrc) {
  const uint32_t ldsoff = (uint32_t)(uintptr_t)lds_dst;
  asm volatile("global_load_async_to_lds_b128 %0, %1, off"
               :: "v"(ldsoff), "v"(gsrc)
               : "memory");
}

static __device__ __forceinline__ void async_wait0() {
  asm volatile("s_wait_asynccnt 0" ::: "memory");
}

// --------------------------- weight prep -----------------------------------

__global__ __launch_bounds__(256) void wprep_kernel(
    const float* __restrict__ W_enc,
    const float* __restrict__ w_ih0, const float* __restrict__ w_hh0,
    const float* __restrict__ w_ih1, const float* __restrict__ w_hh1,
    const float* __restrict__ W_pol, const float* __restrict__ W_base,
    const float* __restrict__ b_ih0, const float* __restrict__ b_hh0,
    const float* __restrict__ b_ih1, const float* __restrict__ b_hh1,
    const float* __restrict__ b_pol, const float* __restrict__ b_base,
    h16* __restrict__ wenc16,
    h16* __restrict__ wih0p, h16* __restrict__ whh0p,
    h16* __restrict__ wih1p, h16* __restrict__ whh1p,
    h16* __restrict__ wcomb,
    float* __restrict__ bias0, float* __restrict__ bias1,
    float* __restrict__ hbias)
{
  const size_t NENC = (size_t)512 * 512;
  const size_t NW   = (size_t)G_D * HP;
  const size_t NCMB = (size_t)16 * HP;
  const size_t total = NENC + 4 * NW + NCMB + 2 * G_D + 16;
  const size_t stride = (size_t)gridDim.x * blockDim.x;
  for (size_t i = (size_t)blockIdx.x * blockDim.x + threadIdx.x; i < total; i += stride) {
    if (i < NENC) { wenc16[i] = (h16)W_enc[i]; continue; }
    size_t j = i - NENC;
    if (j < 4 * NW) {
      const int sel = (int)(j / NW);
      const size_t r = (j % NW) / HP, k = (j % NW) % HP;
      const float* src = sel == 0 ? w_ih0 : sel == 1 ? w_hh0 : sel == 2 ? w_ih1 : w_hh1;
      h16* dst = sel == 0 ? wih0p : sel == 1 ? whh0p : sel == 2 ? wih1p : whh1p;
      dst[r * HP + k] = (k < H_D) ? (h16)src[r * H_D + k] : (h16)0.f;
      continue;
    }
    j -= 4 * NW;
    if (j < NCMB) {
      const size_t r = j / HP, k = j % HP;
      float v = 0.f;
      if (k < H_D) v = (r < 15) ? W_pol[r * H_D + k] : W_base[k];
      wcomb[j] = (h16)v;
      continue;
    }
    j -= NCMB;
    if (j < G_D) { bias0[j] = b_ih0[j] + b_hh0[j]; continue; }
    j -= G_D;
    if (j < G_D) { bias1[j] = b_ih1[j] + b_hh1[j]; continue; }
    j -= G_D;
    hbias[j] = (j < 15) ? b_pol[j] : b_base[0];
  }
}

// x[:, 512..543]: clipped reward, one_hot(action), zero pad. Also nd mask.
__global__ __launch_bounds__(256) void pack_tail_kernel(
    const float* __restrict__ reward, const int* __restrict__ act,
    const unsigned char* __restrict__ term,
    h16* __restrict__ x16, float* __restrict__ ndv)
{
  const int row = blockIdx.x * 256 + threadIdx.x;
  if (row >= TBROWS) return;
  const float r = fminf(fmaxf(reward[row], -1.f), 1.f);
  const int a = act[row];
  h16* xr = x16 + (size_t)row * HP;
  xr[FEAT_D] = (h16)r;
  #pragma unroll
  for (int j = 0; j < NA; ++j) xr[FEAT_D + 1 + j] = (h16)((j == a) ? 1.f : 0.f);
  #pragma unroll
  for (int j = H_D; j < HP; ++j) xr[j] = (h16)0.f;
  ndv[row] = term[row] ? 0.f : 1.f;
}

// ------------------- encoder: x[:, :512] = relu(obs @ Wenc^T + b) -----------
// macro tile 128x64, 8 waves (4x2), each wave 32x32 (4 WMMAs / K-step)
// A (f32 obs) is converted during staging; B (f16 weights) goes async->LDS.

__global__ __launch_bounds__(256) void encoder_kernel(
    const float* __restrict__ obs, const h16* __restrict__ Wf,
    const float* __restrict__ benc, h16* __restrict__ x16)
{
  __shared__ h16 As[128 * LDA];
  __shared__ h16 Bs[64 * LDA];
  const int bx = blockIdx.x;
  const int nt = bx & 7, mt = bx >> 3;
  const int mbase = mt * 128, nbase = nt * 64;
  const int tid = threadIdx.x, lane = tid & 31, w = tid >> 5;
  const int wm = w & 3, wn = w >> 2;
  const int ar = tid >> 1, aoff = (tid & 1) * 16;
  const int bc = tid >> 2, boff = (tid & 3) * 8;
  v8f acc[2][2] = {};
  for (int k0 = 0; k0 < OBS_D; k0 += 32) {
    async_cp16(Bs + bc * LDA + boff,
               Wf + (size_t)(nbase + bc) * OBS_D + k0 + boff);
    const float* asrc = obs + (size_t)(mbase + ar) * OBS_D + k0 + aoff;
    h16* adst = As + ar * LDA + aoff;
    #pragma unroll
    for (int i = 0; i < 16; i += 4) {
      v4f v = *(const v4f*)(asrc + i);
      adst[i + 0] = (h16)v[0]; adst[i + 1] = (h16)v[1];
      adst[i + 2] = (h16)v[2]; adst[i + 3] = (h16)v[3];
    }
    async_wait0();
    __syncthreads();
    v16h a0 = frag_a(As, wm * 32, lane);
    v16h a1 = frag_a(As, wm * 32 + 16, lane);
    v16h b0 = frag_b(Bs, wn * 32, lane);
    v16h b1 = frag_b(Bs, wn * 32 + 16, lane);
    acc[0][0] = wmma_f16(a0, b0, acc[0][0]);
    acc[0][1] = wmma_f16(a0, b1, acc[0][1]);
    acc[1][0] = wmma_f16(a1, b0, acc[1][0]);
    acc[1][1] = wmma_f16(a1, b1, acc[1][1]);
    __syncthreads();
  }
  const int ln = lane & 15, hs = lane >> 4;
  #pragma unroll
  for (int mi = 0; mi < 2; ++mi)
    #pragma unroll
    for (int ni = 0; ni < 2; ++ni) {
      const int col = nbase + wn * 32 + ni * 16 + ln;
      const float bv = benc[col];
      #pragma unroll
      for (int v = 0; v < 8; ++v) {
        const int r = mbase + wm * 32 + mi * 16 + hs * 8 + v;
        x16[(size_t)r * HP + col] = (h16)fmaxf(acc[mi][ni][v] + bv, 0.f);
      }
    }
}

// -------- pre0 = x @ w_ih0^T + (b_ih0 + b_hh0), all T at once (f16) --------
// Both A and B staged with async global->LDS copies.

__global__ __launch_bounds__(256) void preih0_kernel(
    const h16* __restrict__ x16, const h16* __restrict__ W,
    const float* __restrict__ bias0, h16* __restrict__ pre0)
{
  __shared__ h16 As[128 * LDA];
  __shared__ h16 Bs[64 * LDA];
  const int bx = blockIdx.x;
  const int mt = bx / 33, nt = bx % 33;
  const int mbase = mt * 128, nbase = nt * 64;
  const int tid = threadIdx.x, lane = tid & 31, w = tid >> 5;
  const int wm = w & 3, wn = w >> 2;
  const int ar = tid >> 1, aoff = (tid & 1) * 16;
  const int bc = tid >> 2, boff = (tid & 3) * 8;
  v8f acc[2][2] = {};
  for (int k0 = 0; k0 < HP; k0 += 32) {
    const h16* asrc = x16 + (size_t)(mbase + ar) * HP + k0 + aoff;
    h16* adst = As + ar * LDA + aoff;
    async_cp16(adst, asrc);
    async_cp16(adst + 8, asrc + 8);
    async_cp16(Bs + bc * LDA + boff,
               W + (size_t)(nbase + bc) * HP + k0 + boff);
    async_wait0();
    __syncthreads();
    v16h a0 = frag_a(As, wm * 32, lane);
    v16h a1 = frag_a(As, wm * 32 + 16, lane);
    v16h b0 = frag_b(Bs, wn * 32, lane);
    v16h b1 = frag_b(Bs, wn * 32 + 16, lane);
    acc[0][0] = wmma_f16(a0, b0, acc[0][0]);
    acc[0][1] = wmma_f16(a0, b1, acc[0][1]);
    acc[1][0] = wmma_f16(a1, b0, acc[1][0]);
    acc[1][1] = wmma_f16(a1, b1, acc[1][1]);
    __syncthreads();
  }
  const int ln = lane & 15, hs = lane >> 4;
  #pragma unroll
  for (int mi = 0; mi < 2; ++mi)
    #pragma unroll
    for (int ni = 0; ni < 2; ++ni) {
      const int col = nbase + wn * 32 + ni * 16 + ln;
      const float bv = bias0[col];
      #pragma unroll
      for (int v = 0; v < 8; ++v) {
        const int r = mbase + wm * 32 + mi * 16 + hs * 8 + v;
        pre0[(size_t)r * G_D + col] = (h16)(acc[mi][ni][v] + bv);
      }
    }
}

// ------------------------- sequential LSTM step -----------------------------
// One (layer, t). Macro tile M=128 x N=64, where the 64 N-columns are the
// SAME 16 j-columns of all four gates (i,f,g,o) -> cell fused via LDS.
// The done-mask h*nd is applied to the GEMM RESULT rows (row scale commutes
// with the GEMM), so A staging is a pure f16 copy -> async global->LDS.
// TWO=true (layer 1) runs two K-passes (ih then hh) into separate register
// accumulator sets so each contribution gets its own nd scale.
// Double-buffered LDS tiles: compute buf[s&1] while async-staging the other.
// grid = 2 (M tiles) * 33 (j strips) = 66 workgroups.

template <bool TWO>
__global__ __launch_bounds__(256) void lstm_step_kernel(
    const h16* __restrict__ A1, int nd1,
    const h16* __restrict__ W1,
    const h16* __restrict__ A2, int nd2,
    const h16* __restrict__ W2,            // used only when TWO
    const h16* __restrict__ pre,           // [B][G] (incl. biases) or null
    const float* __restrict__ bias,        // [G] or null
    const float* __restrict__ ndv,         // nd[t] : [B]
    float* __restrict__ c_state,           // [B][HP] f32 (in/out)
    h16* __restrict__ h_out16,             // [B][HP]
    h16* __restrict__ lstm_out_t,          // [B][HP] or null
    float* __restrict__ h_out32)           // [B][HP]
{
  __shared__ h16 As[2][128 * LDA];
  __shared__ h16 Bs[2][64 * LDA];
  __shared__ float gbuf[128 * 64];
  const int bx = blockIdx.x;
  const int mt = bx / 33, jt = bx % 33;
  const int mbase = mt * 128, j0 = jt * 16;
  const int tid = threadIdx.x, lane = tid & 31, w = tid >> 5;
  const int wm = w & 3, wn = w >> 2;
  const int ar = tid >> 1, aoff = (tid & 1) * 16;
  const int bc = tid >> 2, boff = (tid & 3) * 8;
  const int brow = (bc >> 4) * H_D + j0 + (bc & 15);   // gate-grouped W row

  auto stage = [&](int buf, const h16* Ap, const h16* Wp, int k0) {
    const h16* asrc = Ap + (size_t)(mbase + ar) * HP + k0 + aoff;
    h16* adst = As[buf] + ar * LDA + aoff;
    async_cp16(adst, asrc);
    async_cp16(adst + 8, asrc + 8);
    async_cp16(Bs[buf] + bc * LDA + boff,
               Wp + (size_t)brow * HP + k0 + boff);
  };

  constexpr int KT = HP / 32;               // 17
  constexpr int NK = TWO ? 2 * KT : KT;

  v8f acc1[2][2] = {};
  v8f acc2[2][2] = {};

  stage(0, A1, W1, 0);
  async_wait0();
  __syncthreads();

  for (int s = 0; s < NK; ++s) {
    const int buf = s & 1;
    if (s + 1 < NK) {
      const int ns = s + 1;
      const h16* Ap = (TWO && ns >= KT) ? A2 : A1;
      const h16* Wp = (TWO && ns >= KT) ? W2 : W1;
      stage(buf ^ 1, Ap, Wp, (ns % KT) * 32);
    }
    v16h a0 = frag_a(As[buf], wm * 32, lane);
    v16h a1 = frag_a(As[buf], wm * 32 + 16, lane);
    v16h b0 = frag_b(Bs[buf], wn * 32, lane);
    v16h b1 = frag_b(Bs[buf], wn * 32 + 16, lane);
    if (TWO && s >= KT) {
      acc2[0][0] = wmma_f16(a0, b0, acc2[0][0]);
      acc2[0][1] = wmma_f16(a0, b1, acc2[0][1]);
      acc2[1][0] = wmma_f16(a1, b0, acc2[1][0]);
      acc2[1][1] = wmma_f16(a1, b1, acc2[1][1]);
    } else {
      acc1[0][0] = wmma_f16(a0, b0, acc1[0][0]);
      acc1[0][1] = wmma_f16(a0, b1, acc1[0][1]);
      acc1[1][0] = wmma_f16(a1, b0, acc1[1][0]);
      acc1[1][1] = wmma_f16(a1, b1, acc1[1][1]);
    }
    async_wait0();
    __syncthreads();
  }

  // stage gate pre-activations to LDS, applying per-row done-mask scales
  const int ln = lane & 15, hs = lane >> 4;
  #pragma unroll
  for (int mi = 0; mi < 2; ++mi)
    #pragma unroll
    for (int ni = 0; ni < 2; ++ni)
      #pragma unroll
      for (int v = 0; v < 8; ++v) {
        const int row = wm * 32 + mi * 16 + hs * 8 + v;
        const float nd = ndv[mbase + row];
        const float s1 = nd1 ? nd : 1.f;
        float g = acc1[mi][ni][v] * s1;
        if (TWO) {
          const float s2 = nd2 ? nd : 1.f;
          g += acc2[mi][ni][v] * s2;
        }
        gbuf[row * 64 + wn * 32 + ni * 16 + ln] = g;
      }
  __syncthreads();

  // fused LSTM cell (128 rows x 16 j-cols)
  for (int idx = tid; idx < 128 * 16; idx += 256) {
    const int r = idx >> 4, jj = idx & 15;
    const int rg = mbase + r, col = j0 + jj;
    float gi = gbuf[r * 64 + jj];
    float gf = gbuf[r * 64 + 16 + jj];
    float gg = gbuf[r * 64 + 32 + jj];
    float go = gbuf[r * 64 + 48 + jj];
    if (pre) {
      const h16* pp = pre + (size_t)rg * G_D + col;
      gi += (float)pp[0];
      gf += (float)pp[H_D];
      gg += (float)pp[2 * H_D];
      go += (float)pp[3 * H_D];
    }
    if (bias) {
      gi += bias[col];
      gf += bias[H_D + col];
      gg += bias[2 * H_D + col];
      go += bias[3 * H_D + col];
    }
    const float nd = ndv[rg];
    const float cp = c_state[(size_t)rg * HP + col] * nd;
    const float si = 1.f / (1.f + __expf(-gi));
    const float sf = 1.f / (1.f + __expf(-gf));
    const float so = 1.f / (1.f + __expf(-go));
    const float cn = sf * cp + si * tanhf(gg);
    const float hn = so * tanhf(cn);
    c_state[(size_t)rg * HP + col] = cn;
    const h16 hh = (h16)hn;
    h_out16[(size_t)rg * HP + col] = hh;
    if (lstm_out_t) lstm_out_t[(size_t)rg * HP + col] = hh;
    if (h_out32) h_out32[(size_t)rg * HP + col] = hn;
  }
}

// ---------------- heads: [T*B,16] = lstm_out @ Wcomb^T + bias ---------------
// Wcomb rows 0..14 = W_pol, row 15 = W_base. 4 waves, macro 64x16.

__global__ __launch_bounds__(128) void heads_kernel(
    const h16* __restrict__ lo16, const h16* __restrict__ Wc,
    const float* __restrict__ hbias, float* __restrict__ logits)
{
  __shared__ h16 As[64 * LDA];
  __shared__ h16 Bs[16 * LDA];
  const int mbase = blockIdx.x * 64;
  const int tid = threadIdx.x, lane = tid & 31, w = tid >> 5;
  const int ar = tid >> 1, aoff = (tid & 1) * 16;
  v8f acc = {};
  for (int k0 = 0; k0 < HP; k0 += 32) {
    const h16* asrc = lo16 + (size_t)(mbase + ar) * HP + k0 + aoff;
    h16* adst = As + ar * LDA + aoff;
    async_cp16(adst, asrc);
    async_cp16(adst + 8, asrc + 8);
    if (tid < 64) {
      const int bc = tid >> 2, boff = (tid & 3) * 8;
      async_cp16(Bs + bc * LDA + boff,
                 Wc + (size_t)bc * HP + k0 + boff);
    }
    async_wait0();
    __syncthreads();
    v16h a = frag_a(As, w * 16, lane);
    v16h b = frag_b(Bs, 0, lane);
    acc = wmma_f16(a, b, acc);
    __syncthreads();
  }
  const int ln = lane & 15, hs = lane >> 4;
  const float bv = hbias[ln];
  #pragma unroll
  for (int v = 0; v < 8; ++v) {
    const int r = mbase + w * 16 + hs * 8 + v;
    logits[(size_t)r * 16 + ln] = acc[v] + bv;
  }
}

// --------------------------- output emission --------------------------------
// NOTE: jax.random.categorical's PRNG stream is not reproducible here;
// action = deterministic argmax of the policy logits.

__global__ __launch_bounds__(256) void emit_outputs_kernel(
    const float* __restrict__ logits, float* __restrict__ out)
{
  const int row = blockIdx.x * 256 + threadIdx.x;
  if (row >= TBROWS) return;
  const float* lg = logits + (size_t)row * 16;
  float mx = -3.4e38f; int am = 0;
  #pragma unroll
  for (int j = 0; j < NA; ++j) {
    const float v = lg[j];
    out[(size_t)row * NA + j] = v;
    if (v > mx) { mx = v; am = j; }
  }
  out[OFF_BASE + row] = lg[15];
  out[OFF_ACT + row] = (float)am;
}

__global__ __launch_bounds__(256) void emit_state_kernel(
    const float* __restrict__ h0, const float* __restrict__ h1,
    const float* __restrict__ c0, const float* __restrict__ c1,
    float* __restrict__ out)
{
  const int idx = blockIdx.x * 256 + threadIdx.x;
  const int n1 = 2 * B_DIM * H_D;
  if (idx >= 2 * n1) return;
  const int which = idx / n1;           // 0 -> hT, 1 -> cT
  const int rem = idx % n1;
  const int l = rem / (B_DIM * H_D);
  const int b = (rem / H_D) % B_DIM;
  const int j = rem % H_D;
  const float* src = which ? (l ? c1 : c0) : (l ? h1 : h0);
  out[(which ? OFF_CT : OFF_HT) + rem] = src[(size_t)b * HP + j];
}

// ------------------------------- host side ----------------------------------

extern "C" void kernel_launch(void* const* d_in, const int* in_sizes, int n_in,
                              void* d_out, int out_size, void* d_ws, size_t ws_size,
                              hipStream_t stream) {
  (void)in_sizes; (void)n_in; (void)out_size; (void)ws_size;
  const float*         obs    = (const float*)d_in[0];
  const int*           lact   = (const int*)d_in[1];
  const float*         reward = (const float*)d_in[2];
  const unsigned char* term   = (const unsigned char*)d_in[3];
  const float* W_enc  = (const float*)d_in[4];
  const float* b_enc  = (const float*)d_in[5];
  const float* w_ih0  = (const float*)d_in[6];
  const float* w_hh0  = (const float*)d_in[7];
  const float* b_ih0  = (const float*)d_in[8];
  const float* b_hh0  = (const float*)d_in[9];
  const float* w_ih1  = (const float*)d_in[10];
  const float* w_hh1  = (const float*)d_in[11];
  const float* b_ih1  = (const float*)d_in[12];
  const float* b_hh1  = (const float*)d_in[13];
  const float* W_pol  = (const float*)d_in[14];
  const float* b_pol  = (const float*)d_in[15];
  const float* W_base = (const float*)d_in[16];
  const float* b_base = (const float*)d_in[17];
  float* out = (float*)d_out;
  char* ws = (char*)d_ws;

  // workspace layout (256B aligned)
  size_t o = 0;
  auto nxt = [&](size_t bytes) { size_t r = o; o += (bytes + 255) & ~(size_t)255; return r; };
  const size_t o_x16   = nxt((size_t)TBROWS * HP * 2);
  const size_t o_pre0  = nxt((size_t)TBROWS * G_D * 2);
  const size_t o_lout  = nxt((size_t)TBROWS * HP * 2);
  const size_t o_wenc  = nxt((size_t)512 * 512 * 2);
  const size_t o_wih0  = nxt((size_t)G_D * HP * 2);
  const size_t o_whh0  = nxt((size_t)G_D * HP * 2);
  const size_t o_wih1  = nxt((size_t)G_D * HP * 2);
  const size_t o_whh1  = nxt((size_t)G_D * HP * 2);
  const size_t o_wcomb = nxt((size_t)16 * HP * 2);
  const size_t o_bias0 = nxt((size_t)G_D * 4);
  const size_t o_bias1 = nxt((size_t)G_D * 4);
  const size_t o_hbias = nxt((size_t)16 * 4);
  const size_t o_ndv   = nxt((size_t)TBROWS * 4);
  const size_t o_h0a   = nxt((size_t)B_DIM * HP * 2);
  const size_t o_h0b   = nxt((size_t)B_DIM * HP * 2);
  const size_t o_h1a   = nxt((size_t)B_DIM * HP * 2);
  const size_t o_h1b   = nxt((size_t)B_DIM * HP * 2);
  const size_t o_c0    = nxt((size_t)B_DIM * HP * 4);
  const size_t o_c1    = nxt((size_t)B_DIM * HP * 4);
  const size_t o_h0f   = nxt((size_t)B_DIM * HP * 4);
  const size_t o_h1f   = nxt((size_t)B_DIM * HP * 4);
  const size_t o_log   = nxt((size_t)TBROWS * 16 * 4);

  h16*  x16    = (h16*)(ws + o_x16);
  h16*  pre0   = (h16*)(ws + o_pre0);
  h16*  lout   = (h16*)(ws + o_lout);
  h16*  wenc16 = (h16*)(ws + o_wenc);
  h16*  wih0p  = (h16*)(ws + o_wih0);
  h16*  whh0p  = (h16*)(ws + o_whh0);
  h16*  wih1p  = (h16*)(ws + o_wih1);
  h16*  whh1p  = (h16*)(ws + o_whh1);
  h16*  wcomb  = (h16*)(ws + o_wcomb);
  float* bias0 = (float*)(ws + o_bias0);
  float* bias1 = (float*)(ws + o_bias1);
  float* hbias = (float*)(ws + o_hbias);
  float* ndv   = (float*)(ws + o_ndv);
  h16*  h016[2] = { (h16*)(ws + o_h0a), (h16*)(ws + o_h0b) };
  h16*  h116[2] = { (h16*)(ws + o_h1a), (h16*)(ws + o_h1b) };
  float* c0    = (float*)(ws + o_c0);
  float* c1    = (float*)(ws + o_c1);
  float* h0f   = (float*)(ws + o_h0f);
  float* h1f   = (float*)(ws + o_h1f);
  float* logits = (float*)(ws + o_log);

  // zero-init state (graph-capturable memset nodes)
  hipMemsetAsync(h016[0], 0, (size_t)B_DIM * HP * 2, stream);
  hipMemsetAsync(h016[1], 0, (size_t)B_DIM * HP * 2, stream);
  hipMemsetAsync(h116[0], 0, (size_t)B_DIM * HP * 2, stream);
  hipMemsetAsync(h116[1], 0, (size_t)B_DIM * HP * 2, stream);
  hipMemsetAsync(c0, 0, (size_t)B_DIM * HP * 4, stream);
  hipMemsetAsync(c1, 0, (size_t)B_DIM * HP * 4, stream);
  hipMemsetAsync(h0f, 0, (size_t)B_DIM * HP * 4, stream);
  hipMemsetAsync(h1f, 0, (size_t)B_DIM * HP * 4, stream);
  hipMemsetAsync(lout, 0, (size_t)TBROWS * HP * 2, stream);  // pad cols must be 0

  wprep_kernel<<<4096, 256, 0, stream>>>(
      W_enc, w_ih0, w_hh0, w_ih1, w_hh1, W_pol, W_base,
      b_ih0, b_hh0, b_ih1, b_hh1, b_pol, b_base,
      wenc16, wih0p, whh0p, wih1p, whh1p, wcomb, bias0, bias1, hbias);

  pack_tail_kernel<<<TBROWS / 256, 256, 0, stream>>>(reward, lact, term, x16, ndv);

  encoder_kernel<<<(TBROWS / 128) * 8, 256, 0, stream>>>(obs, wenc16, b_enc, x16);

  preih0_kernel<<<(TBROWS / 128) * 33, 256, 0, stream>>>(x16, wih0p, bias0, pre0);

  for (int t = 0; t < T_DIM; ++t) {
    const int pa = t & 1, pb = (t + 1) & 1;
    // layer 0: gates = pre0[t] + nd*(h0 @ w_hh0^T)
    lstm_step_kernel<false><<<66, 256, 0, stream>>>(
        h016[pa], 1, whh0p,
        nullptr, 0, nullptr,
        pre0 + (size_t)t * B_DIM * G_D, nullptr,
        ndv + (size_t)t * B_DIM,
        c0, h016[pb], nullptr, h0f);
    // layer 1: gates = h0_new @ w_ih1^T + nd*(h1 @ w_hh1^T) + bias1
    lstm_step_kernel<true><<<66, 256, 0, stream>>>(
        h016[pb], 0, wih1p,
        h116[pa], 1, whh1p,
        nullptr, bias1,
        ndv + (size_t)t * B_DIM,
        c1, h116[pb], lout + (size_t)t * B_DIM * HP, h1f);
  }

  heads_kernel<<<TBROWS / 64, 128, 0, stream>>>(lout, wcomb, hbias, logits);

  emit_outputs_kernel<<<TBROWS / 256, 256, 0, stream>>>(logits, out);
  emit_state_kernel<<<(4 * B_DIM * H_D + 255) / 256, 256, 0, stream>>>(
      h0f, h1f, c0, c1, out);
}
